// ViGNN_32280974197470
// MI455X (gfx1250) — matrix operand order repaired and checked by reading, not compiled, not run on hardware
//
#include <hip/hip_runtime.h>
#include <hip/hip_bf16.h>
#include <math.h>

typedef __attribute__((ext_vector_type(16))) _Float16 v16h;
typedef __attribute__((ext_vector_type(8)))  float    v8f;

// ---------------------------------------------------------------------------
// WMMA skinny GEMM: out[16,N] = X[16,K] @ W[K,:] (+bias). W addressed as
// W[k*LDW + n]; caller pre-offsets W/bias pointers for column slices (e.g.
// the v-part of qkv). One block = 2 waves, each wave owns one 16-col tile.
// K and LDW are compile-time so ALL W loads are base + immediate offset and
// the K loop fully unrolls into a stream of global_load/ds_load/v_wmma.
// A fragment (16x32 f16): lane m=lane&15, hi=lane>>4; K contiguous, K=ks+hi*16+e.
// B fragment (32x16 f16): lanes 0-15 hold K=ks..ks+15, lanes 16-31 K=+16.
// ---------------------------------------------------------------------------
template <int K, int LDW>
__global__ __launch_bounds__(64) void wgemm16_kernel(
    const float* __restrict__ X,
    const float* __restrict__ W,
    const float* __restrict__ bias,
    float* __restrict__ out, int ldo,
    int ntiles, int nvalid) {
  __shared__ _Float16 Xs[16 * K];
  for (int idx = threadIdx.x; idx < 16 * K; idx += 64)
    Xs[idx] = (_Float16)X[idx];
  __syncthreads();

  const int wave = threadIdx.x >> 5;
  const int lane = threadIdx.x & 31;
  const int tile = blockIdx.x * 2 + wave;
  if (tile >= ntiles) return;               // wave-uniform branch (after barrier)

  const int hi = lane >> 4;                 // lane half selects K sub-block
  const int lm = lane & 15;
  const int n  = tile * 16 + lm;
  const int nl = (n < nvalid) ? n : (nvalid - 1);  // clamp OOB cols (padded tile)

  // Single per-lane bases; all subsequent accesses use constant offsets.
  const float*     wp = W + (size_t)(hi << 4) * LDW + nl;   // W[(hi*16 + k)*LDW + nl]
  const _Float16*  xp = Xs + lm * K + (hi << 4);            // Xs[lm*K + hi*16 + ...]

  v8f acc = {};
#pragma unroll
  for (int ks = 0; ks < K; ks += 32) {
    v16h a, b;
#pragma unroll
    for (int e = 0; e < 8; ++e) {           // elements 0..15: K = ks+hi*16+e
      a[e]     = xp[ks + e];                // contiguous 16B -> ds_load_b128
      a[e + 8] = xp[ks + 8 + e];            // contiguous 16B at +16B
    }
#pragma unroll
    for (int e = 0; e < 16; ++e)            // K = ks + hi*16 + e, col nl
      b[e] = (_Float16)wp[(size_t)(ks + e) * LDW];  // base + imm offset
    acc = __builtin_amdgcn_wmma_f32_16x16x32_f16(
        false, a, false, b, (short)0, acc, false, false);
  }

  if (n < nvalid) {
    float bv = bias[nl];
#pragma unroll
    for (int r = 0; r < 8; ++r)             // C layout: vgpr r -> rows r, r+8
      out[(size_t)(r + hi * 8) * ldo + n] = acc[r] + bv;
  }
}

// ---------------------------------------------------------------------------
// In-place LayerNorm (+optional residual add, +optional exact-erf GELU) over
// y[16, N]. One block, 16 waves, one wave per row; N in {256,384,512}.
// ---------------------------------------------------------------------------
template <bool GELU, bool RES>
__global__ __launch_bounds__(512) void ln_kernel(
    float* __restrict__ y, const float* __restrict__ res,
    const float* __restrict__ g, const float* __restrict__ b, int N) {
  const int wave = threadIdx.x >> 5, lane = threadIdx.x & 31;
  const int cnt = N >> 5;                   // <= 16
  float v[16];
  float s = 0.f, ss = 0.f;
  for (int it = 0; it < cnt; ++it) {
    int j = lane + (it << 5);
    float x = y[(size_t)wave * N + j];
    if (RES) x += res[(size_t)wave * N + j];
    v[it] = x; s += x; ss += x * x;
  }
#pragma unroll
  for (int o = 16; o > 0; o >>= 1) {
    s  += __shfl_xor(s,  o, 32);
    ss += __shfl_xor(ss, o, 32);
  }
  float mean = s / N;
  float var  = ss / N - mean * mean;
  float rstd = rsqrtf(var + 1e-5f);
  for (int it = 0; it < cnt; ++it) {
    int j = lane + (it << 5);
    float t = (v[it] - mean) * rstd * g[j] + b[j];
    if (GELU) t = 0.5f * t * (1.0f + erff(t * 0.70710678118654752f));
    y[(size_t)wave * N + j] = t;
  }
}

// ---------------------------------------------------------------------------
// Disease branch (batch-independent, ~40 MFLOP once): plain VALU kernels.
// ---------------------------------------------------------------------------
__global__ void dqkv_kernel(const float* __restrict__ dq,
                            const float* __restrict__ w,
                            const float* __restrict__ bias,
                            float* __restrict__ out) {
  int idx = blockIdx.x * blockDim.x + threadIdx.x;
  if (idx >= 45 * 1152) return;
  int i = idx / 1152, nn = idx - i * 1152;
  float s = bias[nn];
  for (int k = 0; k < 384; ++k) s = fmaf(dq[i * 384 + k], w[(size_t)k * 1152 + nn], s);
  out[idx] = s;
}

__global__ __launch_bounds__(1024) void d_attn_kernel(
    const float* __restrict__ qkvd, float* __restrict__ attno) {
  __shared__ float S[4 * 45 * 45];
  const int tid = threadIdx.x;
  for (int t = tid; t < 4 * 45 * 45; t += 1024) {
    int h = t / 2025, rem = t - h * 2025, i = rem / 45, j = rem - (rem / 45) * 45;
    const float* q = qkvd + (size_t)i * 1152 + h * 96;
    const float* k = qkvd + (size_t)j * 1152 + 384 + h * 96;
    float s = 0.f;
    for (int d = 0; d < 96; ++d) s = fmaf(q[d], k[d], s);
    S[t] = s * 0.10206207262f;              // 1/sqrt(96)
  }
  __syncthreads();
  for (int r = tid; r < 4 * 45; r += 1024) {
    float* row = S + r * 45;
    float mx = row[0];
    for (int j = 1; j < 45; ++j) mx = fmaxf(mx, row[j]);
    float sum = 0.f;
    for (int j = 0; j < 45; ++j) { float e = expf(row[j] - mx); row[j] = e; sum += e; }
    float inv = 1.f / sum;
    for (int j = 0; j < 45; ++j) row[j] *= inv;
  }
  __syncthreads();
  for (int t = tid; t < 45 * 384; t += 1024) {
    int i = t / 384, c = t - i * 384, h = c / 96, d = c - h * 96;
    const float* a = S + (h * 45 + i) * 45;
    float s = 0.f;
    for (int j = 0; j < 45; ++j)
      s = fmaf(a[j], qkvd[(size_t)j * 1152 + 768 + h * 96 + d], s);
    attno[t] = s;
  }
}

__global__ __launch_bounds__(384) void d_final_kernel(
    const float* __restrict__ attno, const float* __restrict__ pw,
    const float* __restrict__ pb, float* __restrict__ daware) {
  __shared__ float mr[384];
  int n = threadIdx.x;
  float s = 0.f;
  for (int i = 0; i < 45; ++i) s += attno[(size_t)i * 384 + n];
  mr[n] = s * (1.0f / 45.0f);
  __syncthreads();
  float o = pb[n];
  for (int k = 0; k < 384; ++k) o = fmaf(mr[k], pw[(size_t)k * 384 + n], o);
  daware[n] = o;
}

__global__ void concat_kernel(const float* __restrict__ gc,
                              const float* __restrict__ dawr,
                              float* __restrict__ f) {
  int idx = blockIdx.x * blockDim.x + threadIdx.x;
  if (idx >= 16 * 768) return;
  int b = idx / 768, j = idx - b * 768;
  f[idx] = (j < 384) ? gc[b * 384 + j] : dawr[j - 384];
}

// ---------------------------------------------------------------------------
extern "C" void kernel_launch(void* const* d_in, const int* /*in_sizes*/, int /*n_in*/,
                              void* d_out, int /*out_size*/, void* d_ws, size_t /*ws_size*/,
                              hipStream_t stream) {
  const float* x_feat = (const float*)d_in[0];
  const float* pp_w = (const float*)d_in[1];
  const float* pp_b = (const float*)d_in[2];
  const float* pp_g = (const float*)d_in[3];
  const float* pp_be = (const float*)d_in[4];
  const float* dq = (const float*)d_in[5];
  const float* da_qkv_w = (const float*)d_in[6];
  const float* da_qkv_b = (const float*)d_in[7];
  const float* da_proj_w = (const float*)d_in[8];
  const float* da_proj_b = (const float*)d_in[9];
  const float* gc_w = (const float*)d_in[10];
  const float* gc_b = (const float*)d_in[11];
  const float* gc_g = (const float*)d_in[12];
  const float* gc_be = (const float*)d_in[13];
  const float* c_w1 = (const float*)d_in[14];
  const float* c_b1 = (const float*)d_in[15];
  const float* c_g1 = (const float*)d_in[16];
  const float* c_be1 = (const float*)d_in[17];
  const float* c_w2 = (const float*)d_in[18];
  const float* c_b2 = (const float*)d_in[19];
  const float* c_g2 = (const float*)d_in[20];
  const float* c_be2 = (const float*)d_in[21];
  const float* c_w3 = (const float*)d_in[22];
  const float* c_b3 = (const float*)d_in[23];

  float* ws = (float*)d_ws;
  float* u0   = ws;               // 16*384
  float* u1   = u0 + 16 * 384;    // 16*384
  float* tbuf = u1 + 16 * 384;    // 16*384
  float* gcb  = tbuf + 16 * 384;  // 16*384
  float* qkvd = gcb + 16 * 384;   // 45*1152
  float* attno = qkvd + 45 * 1152;  // 45*384
  float* dawr = attno + 45 * 384; // 384
  float* fbuf = dawr + 384;       // 16*768
  float* h1   = fbuf + 16 * 768;  // 16*512
  float* h2   = h1 + 16 * 512;    // 16*256

  // 1) u0 = gelu(LN(x_feat @ pp_w + pp_b))   (patch path collapses: NP rows identical)
  wgemm16_kernel<384, 384><<<12, 64, 0, stream>>>(x_feat, pp_w, pp_b, u0, 384, 24, 384);
  ln_kernel<true, false><<<1, 512, 0, stream>>>(u0, nullptr, pp_g, pp_be, 384);

  // 2) three attention layers; identical tokens => attn output == v slice of qkv
  float* bufs[2] = {u0, u1};
  int ci = 0;
  for (int i = 0; i < 3; ++i) {
    const float* qkvw = (const float*)d_in[24 + 6 * i + 0];
    const float* qkvb = (const float*)d_in[24 + 6 * i + 1];
    const float* pjw  = (const float*)d_in[24 + 6 * i + 2];
    const float* pjb  = (const float*)d_in[24 + 6 * i + 3];
    const float* lng  = (const float*)d_in[24 + 6 * i + 4];
    const float* lnb  = (const float*)d_in[24 + 6 * i + 5];
    float* cur = bufs[ci];
    float* nxt = bufs[ci ^ 1];
    // t = cur @ qkv_w[:, 2D:3D] + qkv_b[2D:3D]   (v-slice only)
    wgemm16_kernel<384, 1152><<<12, 64, 0, stream>>>(cur, qkvw + 768, qkvb + 768,
                                                     tbuf, 384, 24, 384);
    // nxt = LN(cur + (t @ proj_w + proj_b))
    wgemm16_kernel<384, 384><<<12, 64, 0, stream>>>(tbuf, pjw, pjb, nxt, 384, 24, 384);
    ln_kernel<false, true><<<1, 512, 0, stream>>>(nxt, cur, lng, lnb, 384);
    ci ^= 1;
  }
  float* ufin = bufs[ci];

  // 3) global_context = gelu(LN(u @ gc_w + gc_b))  (mean over identical patches == row)
  wgemm16_kernel<384, 384><<<12, 64, 0, stream>>>(ufin, gc_w, gc_b, gcb, 384, 24, 384);
  ln_kernel<true, false><<<1, 512, 0, stream>>>(gcb, nullptr, gc_g, gc_be, 384);

  // 4) disease branch (batch-independent): dense 45-token attention, proj, mean
  dqkv_kernel<<<(45 * 1152 + 255) / 256, 256, 0, stream>>>(dq, da_qkv_w, da_qkv_b, qkvd);
  d_attn_kernel<<<1, 1024, 0, stream>>>(qkvd, attno);
  d_final_kernel<<<1, 384, 0, stream>>>(attno, da_proj_w, da_proj_b, dawr);

  // 5) f = [global_context | disease_aware]
  concat_kernel<<<(16 * 768 + 255) / 256, 256, 0, stream>>>(gcb, dawr, fbuf);

  // 6) classifier MLP
  wgemm16_kernel<768, 512><<<16, 64, 0, stream>>>(fbuf, c_w1, c_b1, h1, 512, 32, 512);
  ln_kernel<true, false><<<1, 512, 0, stream>>>(h1, nullptr, c_g1, c_be1, 512);
  wgemm16_kernel<512, 256><<<8, 64, 0, stream>>>(h1, c_w2, c_b2, h2, 256, 16, 256);
  ln_kernel<true, false><<<1, 512, 0, stream>>>(h2, nullptr, c_g2, c_be2, 256);
  // final: N=45 -> 3 tiles (padded to 48, cols clamped/guarded)
  wgemm16_kernel<256, 45><<<2, 64, 0, stream>>>(h2, c_w3, c_b3, (float*)d_out, 45, 3, 45);
}
